// Net_13091060318414
// MI455X (gfx1250) — compile-verified, matrix operand order checked by reading, not run on hardware
//
#include <hip/hip_runtime.h>
#include <hip/hip_bf16.h>

// Problem constants (fixed by the reference)
#define NN 50000          // nodes
#define NE 800000         // edges (before self loops)
#define CD 128            // channels
#define HH 4              // heads
#define DD 32             // head dim
#define GG 128            // graphs
#define ETOT (NE + NN)    // edges + self loops
#define NEG_SLOPE 0.2f
#define EPSF 1e-16f
#define ENC_NEG_INF 0x007FFFFFu   // order-preserving encoding of -inf

typedef __attribute__((ext_vector_type(16))) _Float16 v16h;
typedef __attribute__((ext_vector_type(8)))  _Float16 v8h;
typedef __attribute__((ext_vector_type(8)))  float    v8f;
typedef int v4i_ __attribute__((vector_size(16)));   // matches builtin param type

// CDNA5 async global->LDS path (ASYNCcnt tracked), guarded so it compiles on
// both toolchains and never reaches the host pass.
#if defined(__gfx1250__) &&                                            \
    __has_builtin(__builtin_amdgcn_global_load_async_to_lds_b128) &&   \
    __has_builtin(__builtin_amdgcn_s_wait_asynccnt)
#define HAVE_ASYNC_LDS 1
#else
#define HAVE_ASYNC_LDS 0
#endif

// ---- order-preserving float <-> uint encoding for atomicMax on floats ----
__device__ __forceinline__ unsigned enc_f32(float f) {
  unsigned u = __float_as_uint(f);
  return (u & 0x80000000u) ? ~u : (u | 0x80000000u);
}
__device__ __forceinline__ float dec_f32(unsigned u) {
  return __uint_as_float((u & 0x80000000u) ? (u & 0x7FFFFFFFu) : ~u);
}

// =====================================================================
// 0) one-time transpose+convert of W: Wt[n][k] = (f16) W[k][n]
//    (N-major layout makes every WMMA B-fragment two contiguous 16B chunks)
// =====================================================================
__global__ void k_prep(const float* __restrict__ W, _Float16* __restrict__ Wt) {
  const int t = blockIdx.x * blockDim.x + threadIdx.x;   // 16384 threads
  const int k = t >> 7, n = t & 127;                     // coalesced read of W
  Wt[n * CD + k] = (_Float16)W[t];
}

// =====================================================================
// 1) h = x @ W via v_wmma_f32_16x16x32_f16.
//    grid = 3125 M-tiles, block = 256 threads = 8 waves (one N-tile each).
//    LDS: Wt staged via async b128 copies; x tile staged as f16.
// =====================================================================
__global__ void k_gemm(const float* __restrict__ x,
                       const _Float16* __restrict__ Wt,
                       float* __restrict__ hbuf) {
  __shared__ __align__(16) _Float16 Wl[CD * CD];   // 32 KB, N-major
  __shared__ __align__(16) _Float16 Xh[16 * CD];   // 4 KB, this block's x tile
  const int tid   = threadIdx.x;
  const int mtile = blockIdx.x;

  // ---- stage Wt (32 KB linear) into LDS: 2048 16-byte chunks ----
#if HAVE_ASYNC_LDS
#pragma unroll
  for (int i = 0; i < 8; ++i) {
    const int chunk = tid + i * 256;
    __builtin_amdgcn_global_load_async_to_lds_b128(
        (__attribute__((address_space(1))) v4i_*)((char*)Wt + chunk * 16),
        (__attribute__((address_space(3))) v4i_*)((char*)Wl + chunk * 16),
        0, 0);
  }
#else
  {
    const uint4* gsrc = (const uint4*)Wt;
    uint4* ldst = (uint4*)Wl;
#pragma unroll
    for (int i = 0; i < 8; ++i) ldst[tid + i * 256] = gsrc[tid + i * 256];
  }
#endif

  // ---- stage this block's 16x128 x tile as f16 (shared by all 8 waves) ----
  {
    const int row = tid >> 4, col = (tid & 15) * 8;    // 8 floats per thread
    const float* xr = x + (size_t)(mtile * 16 + row) * CD + col;
    __builtin_prefetch(xr, 0, 0);                      // global_prefetch_b8
    v8h hx;
#pragma unroll
    for (int j = 0; j < 8; ++j) hx[j] = (_Float16)xr[j];
    *(v8h*)&Xh[row * CD + col] = hx;
  }

#if HAVE_ASYNC_LDS
  __builtin_amdgcn_s_wait_asynccnt(0);
#endif
  __syncthreads();

  const int wave  = tid >> 5;        // N tile 0..7
  const int lane  = tid & 31;
  const int mrow  = lane & 15;       // A row / D col index
  const int khalf = lane >> 4;       // 0: K 0-7/16-23, 1: K 8-15/24-31

  const _Float16* arow = &Xh[mrow * CD];
  const _Float16* bcol = &Wl[(wave * 16 + mrow) * CD];

  v8f acc = {};
#pragma unroll
  for (int k0 = 0; k0 < CD; k0 += 32) {
    const int ka = k0 + khalf * 8;
    v8h alo = *(const v8h*)(arow + ka);
    v8h ahi = *(const v8h*)(arow + ka + 16);
    v8h blo = *(const v8h*)(bcol + ka);
    v8h bhi = *(const v8h*)(bcol + ka + 16);
    v16h a = __builtin_shufflevector(alo, ahi, 0, 1, 2, 3, 4, 5, 6, 7,
                                     8, 9, 10, 11, 12, 13, 14, 15);
    v16h b = __builtin_shufflevector(blo, bhi, 0, 1, 2, 3, 4, 5, 6, 7,
                                     8, 9, 10, 11, 12, 13, 14, 15);
    // 8 args: (neg_a, A, neg_b, B, c_mod, C, reuse_a, reuse_b)
    acc = __builtin_amdgcn_wmma_f32_16x16x32_f16(false, a, false, b,
                                                 (short)0, acc, false, false);
  }
  // C/D layout: VGPR v -> row v + 8*khalf, col = lane&15
  float* __restrict__ orow =
      hbuf + (size_t)(mtile * 16 + khalf * 8) * CD + wave * 16 + mrow;
#pragma unroll
  for (int v = 0; v < 8; ++v) orow[(size_t)v * CD] = acc[v];
}

// =====================================================================
// 2) per-node attention logits + init of per-node accumulators
// =====================================================================
__global__ void k_attn(const float* __restrict__ hbuf,
                       const float* __restrict__ att_src,
                       const float* __restrict__ att_dst,
                       float* __restrict__ a_src, float* __restrict__ a_dst,
                       unsigned* __restrict__ mbuf, float* __restrict__ denom,
                       float* __restrict__ Sbuf) {
  const int n = blockIdx.x * blockDim.x + threadIdx.x;
  if (n >= NN) return;
  const float* hr = hbuf + (size_t)n * CD;
  float4* S4 = (float4*)(Sbuf + (size_t)n * CD);
  for (int c = 0; c < CD / 4; ++c) S4[c] = make_float4(0.f, 0.f, 0.f, 0.f);
  for (int hd = 0; hd < HH; ++hd) {
    float s = 0.0f, d = 0.0f;
    for (int k = 0; k < DD; ++k) {
      float hv = hr[hd * DD + k];
      s += hv * att_src[hd * DD + k];
      d += hv * att_dst[hd * DD + k];
    }
    a_src[n * HH + hd] = s;
    a_dst[n * HH + hd] = d;
    mbuf[n * HH + hd]  = ENC_NEG_INF;
    denom[n * HH + hd] = 0.0f;
  }
}

// init per-graph accumulators
__global__ void k_ginit(float* __restrict__ pooled, float* __restrict__ gden,
                        unsigned* __restrict__ gmax) {
  const int i = blockIdx.x * blockDim.x + threadIdx.x;
  if (i < GG * CD) pooled[i] = 0.0f;
  if (i < GG) { gden[i] = 0.0f; gmax[i] = ENC_NEG_INF; }
}

// =====================================================================
// 3) segment max of leaky-relu logits over in-edges of each dst node
// =====================================================================
__global__ void k_edge_max(const int* __restrict__ ei,
                           const float* __restrict__ a_src,
                           const float* __restrict__ a_dst,
                           unsigned* __restrict__ mbuf) {
  const int e = blockIdx.x * blockDim.x + threadIdx.x;
  if (e >= ETOT) return;
  const int s = (e < NE) ? ei[e]      : (e - NE);
  const int d = (e < NE) ? ei[NE + e] : (e - NE);
  for (int hd = 0; hd < HH; ++hd) {
    float v = a_src[s * HH + hd] + a_dst[d * HH + hd];
    v = (v >= 0.0f) ? v : NEG_SLOPE * v;
    atomicMax(&mbuf[d * HH + hd], enc_f32(v));
  }
}

// =====================================================================
// 4) exp-sum + unnormalized weighted aggregation.
//    one thread per (edge, head): 1 denom atomic + 32 feature atomics.
// =====================================================================
__global__ void k_edge_acc(const int* __restrict__ ei,
                           const float* __restrict__ a_src,
                           const float* __restrict__ a_dst,
                           const unsigned* __restrict__ mbuf,
                           const float* __restrict__ hbuf,
                           float* __restrict__ denom, float* __restrict__ Sbuf) {
  const int t = blockIdx.x * blockDim.x + threadIdx.x;
  const int e = t >> 2, hd = t & 3;
  if (e >= ETOT) return;
  const int s = (e < NE) ? ei[e]      : (e - NE);
  const int d = (e < NE) ? ei[NE + e] : (e - NE);
  float v = a_src[s * HH + hd] + a_dst[d * HH + hd];
  v = (v >= 0.0f) ? v : NEG_SLOPE * v;
  const float w = expf(v - dec_f32(mbuf[d * HH + hd]));
  atomicAdd(&denom[d * HH + hd], w);
  const float* hs = hbuf + (size_t)s * CD + hd * DD;
  float* Sd       = Sbuf + (size_t)d * CD + hd * DD;
  for (int k = 0; k < DD; ++k) atomicAdd(&Sd[k], w * hs[k]);
}

// =====================================================================
// 5) normalize, +bias, ELU (in place into Sbuf), gate logit, graph max
// =====================================================================
__global__ void k_node(const float* __restrict__ denom,
                       const float* __restrict__ bias,
                       const float* __restrict__ gate_w,
                       const float* __restrict__ gate_b,
                       const int* __restrict__ batch,
                       float* __restrict__ Sbuf, float* __restrict__ gate,
                       unsigned* __restrict__ gmax) {
  const int n = blockIdx.x * blockDim.x + threadIdx.x;
  if (n >= NN) return;
  float* r = Sbuf + (size_t)n * CD;
  float g = gate_b[0];
  for (int c = 0; c < CD; ++c) {
    float o = r[c] / (denom[n * HH + (c >> 5)] + EPSF) + bias[c];
    o = (o > 0.0f) ? o : (expf(o) - 1.0f);   // ELU
    r[c] = o;
    g += o * gate_w[c];
  }
  gate[n] = g;
  atomicMax(&gmax[batch[n]], enc_f32(g));
}

// =====================================================================
// 6) graph-level softmax-weighted pooling (unnormalized accumulation)
// =====================================================================
__global__ void k_pool(const int* __restrict__ batch,
                       const float* __restrict__ gate,
                       const unsigned* __restrict__ gmax,
                       const float* __restrict__ Sbuf,
                       float* __restrict__ gden, float* __restrict__ pooled) {
  const int n = blockIdx.x * blockDim.x + threadIdx.x;
  if (n >= NN) return;
  const int g = batch[n];
  const float ge = expf(gate[n] - dec_f32(gmax[g]));
  atomicAdd(&gden[g], ge);
  const float* r = Sbuf + (size_t)n * CD;
  float* p = pooled + (size_t)g * CD;
  for (int c = 0; c < CD; ++c) atomicAdd(&p[c], ge * r[c]);
}

// =====================================================================
// 7) tiny MLP head, one thread per graph
// =====================================================================
__global__ void k_head(const float* __restrict__ pooled,
                       const float* __restrict__ gden,
                       const float* __restrict__ w1, const float* __restrict__ b1,
                       const float* __restrict__ w2, const float* __restrict__ b2,
                       float* __restrict__ out) {
  const int g = threadIdx.x;
  if (g >= GG) return;
  float p[CD];
  const float inv = 1.0f / (gden[g] + EPSF);
  for (int c = 0; c < CD; ++c) p[c] = pooled[(size_t)g * CD + c] * inv;
  float acc = b2[0];
  for (int j = 0; j < 50; ++j) {
    float s = b1[j];
    for (int c = 0; c < CD; ++c) s += p[c] * w1[c * 50 + j];
    s = (s > 0.0f) ? s : 0.0f;
    acc += s * w2[j];
  }
  out[g] = acc;
}

extern "C" void kernel_launch(void* const* d_in, const int* in_sizes, int n_in,
                              void* d_out, int out_size, void* d_ws, size_t ws_size,
                              hipStream_t stream) {
  const float* x        = (const float*)d_in[0];
  const int*   ei       = (const int*)  d_in[1];
  const int*   batch    = (const int*)  d_in[2];
  /* d_in[3] = num_graphs (compile-time GG) */
  const float* W        = (const float*)d_in[4];
  const float* att_src  = (const float*)d_in[5];
  const float* att_dst  = (const float*)d_in[6];
  const float* bias     = (const float*)d_in[7];
  const float* gate_w   = (const float*)d_in[8];
  const float* gate_b   = (const float*)d_in[9];
  const float* w1       = (const float*)d_in[10];
  const float* b1       = (const float*)d_in[11];
  const float* w2       = (const float*)d_in[12];
  const float* b2       = (const float*)d_in[13];
  float* out = (float*)d_out;

  float* ws = (float*)d_ws;
  size_t o = 0;
  float*    hbuf   = ws + o;            o += (size_t)NN * CD;   // h = x@W
  float*    Sbuf   = ws + o;            o += (size_t)NN * CD;   // agg, then x2
  float*    a_src  = ws + o;            o += (size_t)NN * HH;
  float*    a_dst  = ws + o;            o += (size_t)NN * HH;
  unsigned* mbuf   = (unsigned*)(ws+o); o += (size_t)NN * HH;
  float*    denom  = ws + o;            o += (size_t)NN * HH;
  float*    gate   = ws + o;            o += (size_t)NN;
  unsigned* gmax   = (unsigned*)(ws+o); o += (size_t)GG;
  float*    gden   = ws + o;            o += (size_t)GG;
  float*    pooled = ws + o;            o += (size_t)GG * CD;
  _Float16* Wt16   = (_Float16*)(ws+o); o += (size_t)CD * CD / 2;

  k_prep<<<(CD * CD) / 256, 256, 0, stream>>>(W, Wt16);
  k_gemm<<<NN / 16, 256, 0, stream>>>(x, Wt16, hbuf);
  k_attn<<<(NN + 255) / 256, 256, 0, stream>>>(hbuf, att_src, att_dst,
                                               a_src, a_dst, mbuf, denom, Sbuf);
  k_ginit<<<(GG * CD + 255) / 256, 256, 0, stream>>>(pooled, gden, gmax);
  k_edge_max<<<(ETOT + 255) / 256, 256, 0, stream>>>(ei, a_src, a_dst, mbuf);
  k_edge_acc<<<(ETOT * HH + 255) / 256, 256, 0, stream>>>(ei, a_src, a_dst,
                                                          mbuf, hbuf, denom, Sbuf);
  k_node<<<(NN + 255) / 256, 256, 0, stream>>>(denom, bias, gate_w, gate_b,
                                               batch, Sbuf, gate, gmax);
  k_pool<<<(NN + 255) / 256, 256, 0, stream>>>(batch, gate, gmax, Sbuf, gden, pooled);
  k_head<<<1, 128, 0, stream>>>(pooled, gden, w1, b1, w2, b2, out);
}
